// AttentionHead_68582037782652
// MI455X (gfx1250) — matrix operand order repaired and checked by reading, not compiled, not run on hardware
//
#include <hip/hip_runtime.h>
#include <hip/hip_bf16.h>
#include <math.h>

typedef __attribute__((ext_vector_type(8)))  float  v8f;
typedef __attribute__((ext_vector_type(4)))  float  f32x4;
typedef __attribute__((ext_vector_type(16))) __bf16 v16bf;
typedef __attribute__((ext_vector_type(8)))  __bf16 bf16x8;
typedef __attribute__((ext_vector_type(4)))  __bf16 bf16x4;

#define B_DIM 8
#define T_SEQ 2048
#define E_DIM 1024
#define H_DIM 64

// LDS row strides (in bf16 elements) padded to 40 (80 B) -> 16B-aligned b128
// chunks and bank-spread row starts (20*r mod 64 is a full-period pattern).
#define XS_STRIDE 40
#define WT_STRIDE 40
#define PL_STRIDE 40

__device__ __forceinline__ v16bf cat8(bf16x8 lo, bf16x8 hi) {
  return __builtin_shufflevector(lo, hi, 0, 1, 2, 3, 4, 5, 6, 7,
                                 8, 9, 10, 11, 12, 13, 14, 15);
}

__device__ __forceinline__ v8f wmma_bf16(v16bf a, v16bf b, v8f c) {
  // D = A(16x32 bf16) x B(32x16 bf16) + C(16x16 f32)
  return __builtin_amdgcn_wmma_f32_16x16x32_bf16(
      false, a, false, b, (short)0, c, false, false);
}

// ---------------------------------------------------------------------------
// Kernel 1: fused QKV projection.  x[B*T,E] f32  ->  q,k bf16 [B*T,H],
// v transposed bf16 [B,H,T].  One block = 128 rows, 8 waves x 16-row tiles.
// ---------------------------------------------------------------------------
__global__ __launch_bounds__(256) void qkv_proj_kernel(
    const float* __restrict__ x,
    const float* __restrict__ Wq, const float* __restrict__ bq,
    const float* __restrict__ Wk, const float* __restrict__ bk,
    const float* __restrict__ Wv, const float* __restrict__ bv,
    __bf16* __restrict__ qo, __bf16* __restrict__ ko,
    __bf16* __restrict__ vto) {
  __shared__ __bf16 xs[128 * XS_STRIDE];        // x tile, bf16, row-major
  __shared__ __bf16 wt[3][H_DIM * WT_STRIDE];   // W^T tiles: [h][k]

  const int tid  = threadIdx.x;
  const int wave = tid >> 5;
  const int lane = tid & 31;
  const int lh   = lane & 15;
  const int half = lane >> 4;
  const int rbase = blockIdx.x * 128;

  v8f acc[12];
#pragma unroll
  for (int i = 0; i < 12; ++i) acc[i] = v8f{};

  for (int k0 = 0; k0 < E_DIM; k0 += 32) {
    // ---- stage x tile (128x32 f32 -> bf16 LDS) ----
    for (int i = tid; i < 128 * 8; i += 256) {
      const int r  = i >> 3;
      const int c4 = (i & 7) * 4;
      f32x4 xv = *(const f32x4*)(x + (size_t)(rbase + r) * E_DIM + k0 + c4);
      bf16x4 xb = {(__bf16)xv.x, (__bf16)xv.y, (__bf16)xv.z, (__bf16)xv.w};
      *(bf16x4*)(&xs[r * XS_STRIDE + c4]) = xb;
    }
    // ---- stage W tiles transposed: wt[w][h][k] ----
#pragma unroll
    for (int w = 0; w < 3; ++w) {
      const float* W = (w == 0) ? Wq : (w == 1) ? Wk : Wv;
      for (int i = tid; i < 512; i += 256) {
        const int kr = i >> 4;          // 0..31 (k within tile)
        const int c4 = (i & 15) * 4;    // h base
        f32x4 wv4 = *(const f32x4*)(W + (size_t)(k0 + kr) * H_DIM + c4);
        wt[w][(c4 + 0) * WT_STRIDE + kr] = (__bf16)wv4.x;
        wt[w][(c4 + 1) * WT_STRIDE + kr] = (__bf16)wv4.y;
        wt[w][(c4 + 2) * WT_STRIDE + kr] = (__bf16)wv4.z;
        wt[w][(c4 + 3) * WT_STRIDE + kr] = (__bf16)wv4.w;
      }
    }
    __syncthreads();

    // ---- A operand: 16x32 x-tile rows of this wave ----
    const __bf16* xrow = &xs[(wave * 16 + lh) * XS_STRIDE];
    v16bf a = cat8(*(const bf16x8*)(xrow + 16 * half),
                   *(const bf16x8*)(xrow + 16 * half + 8));
#pragma unroll
    for (int w = 0; w < 3; ++w) {
#pragma unroll
      for (int ht = 0; ht < 4; ++ht) {
        const __bf16* wrow = &wt[w][(ht * 16 + lh) * WT_STRIDE + 16 * half];
        v16bf bmat = cat8(*(const bf16x8*)(wrow),
                          *(const bf16x8*)(wrow + 8));
        acc[w * 4 + ht] = wmma_bf16(a, bmat, acc[w * 4 + ht]);
      }
    }
    __syncthreads();
  }

  // ---- epilogue: add bias, cast bf16, store (v transposed) ----
  const int row0 = rbase + wave * 16;
#pragma unroll
  for (int w = 0; w < 3; ++w) {
    const float* bias = (w == 0) ? bq : (w == 1) ? bk : bv;
#pragma unroll
    for (int ht = 0; ht < 4; ++ht) {
      const int h = ht * 16 + lh;
      const float bsc = bias[h];
      v8f c = acc[w * 4 + ht];
#pragma unroll
      for (int r = 0; r < 8; ++r) {
        const int row = row0 + r + 8 * half;   // global token index
        const __bf16 val = (__bf16)(c[r] + bsc);
        if (w == 0) {
          qo[(size_t)row * H_DIM + h] = val;
        } else if (w == 1) {
          ko[(size_t)row * H_DIM + h] = val;
        } else {
          const int bb = row / T_SEQ;
          const int tt = row % T_SEQ;
          vto[((size_t)bb * H_DIM + h) * T_SEQ + tt] = val;
        }
      }
    }
  }
}

// ---------------------------------------------------------------------------
// Kernel 2: causal flash attention.  One wave = 16 query rows, key blocks of
// 32 (= K of bf16 WMMA).  q,k: bf16 [B*T,H]; vt: bf16 [B,H,T]; out f32.
// ---------------------------------------------------------------------------
__global__ __launch_bounds__(256) void flash_attn_kernel(
    const __bf16* __restrict__ q, const __bf16* __restrict__ k,
    const __bf16* __restrict__ vt, float* __restrict__ out) {
  __shared__ __bf16 plds[8][16 * PL_STRIDE];  // per-wave P staging tile

  const int wave = threadIdx.x >> 5;
  const int lane = threadIdx.x & 31;
  const int lh   = lane & 15;
  const int half = lane >> 4;
  const int b    = blockIdx.x >> 4;   // T/128 = 16 query blocks per batch
  const int tb   = blockIdx.x & 15;
  const int qbase = tb * 128 + wave * 16;
  const float scale = 0.125f;         // 1/sqrt(64)

  // Q in A-layout: lane holds row (qbase+lh), K contiguous {16*half..16*half+15}
  const __bf16* qrow = q + (size_t)(b * T_SEQ + qbase + lh) * H_DIM;
  const v16bf a0 = cat8(*(const bf16x8*)(qrow + 16 * half),
                        *(const bf16x8*)(qrow + 16 * half + 8));
  const v16bf a1 = cat8(*(const bf16x8*)(qrow + 32 + 16 * half),
                        *(const bf16x8*)(qrow + 40 + 16 * half));

  float mrow[8], lrow[8];
#pragma unroll
  for (int r = 0; r < 8; ++r) { mrow[r] = -INFINITY; lrow[r] = 0.0f; }
  v8f o[4];
#pragma unroll
  for (int i = 0; i < 4; ++i) o[i] = v8f{};

  __bf16* pl = &plds[wave][0];

  for (int j0 = 0; j0 <= qbase + 15; j0 += 32) {
    // ---- S = Q @ K^T : two 16x16 N-tiles, K(=h) split 0..31 / 32..63 ----
    v8f s[2];
#pragma unroll
    for (int t = 0; t < 2; ++t) {
      const __bf16* krow =
          k + (size_t)(b * T_SEQ + j0 + 16 * t + lh) * H_DIM + 16 * half;
      v16bf bk0 = cat8(*(const bf16x8*)(krow), *(const bf16x8*)(krow + 8));
      v16bf bk1 = cat8(*(const bf16x8*)(krow + 32),
                       *(const bf16x8*)(krow + 40));
      v8f c = wmma_bf16(a0, bk0, v8f{});
      s[t] = wmma_bf16(a1, bk1, c);
    }

    // ---- scale, causal mask, online softmax ----
#pragma unroll
    for (int r = 0; r < 8; ++r) {
      const int qi = qbase + r + 8 * half;
      float sv0 = s[0][r] * scale;
      float sv1 = s[1][r] * scale;
      sv0 = (j0 + lh <= qi) ? sv0 : -INFINITY;
      sv1 = (j0 + 16 + lh <= qi) ? sv1 : -INFINITY;
      float mloc = fmaxf(sv0, sv1);
#pragma unroll
      for (int off = 1; off < 16; off <<= 1)
        mloc = fmaxf(mloc, __shfl_xor(mloc, off, 16));
      const float mnew = fmaxf(mrow[r], mloc);
      const float corr = __expf(mrow[r] - mnew);
      const float p0 = __expf(sv0 - mnew);
      const float p1 = __expf(sv1 - mnew);
      float sum = p0 + p1;
#pragma unroll
      for (int off = 1; off < 16; off <<= 1)
        sum += __shfl_xor(sum, off, 16);
      mrow[r] = mnew;
      lrow[r] = lrow[r] * corr + sum;
#pragma unroll
      for (int ht = 0; ht < 4; ++ht) o[ht][r] *= corr;
      // stage P (C-layout -> LDS row-major) for A-layout reload
      pl[(r + 8 * half) * PL_STRIDE + lh]      = (__bf16)p0;
      pl[(r + 8 * half) * PL_STRIDE + 16 + lh] = (__bf16)p1;
    }

    // ---- reload P in A-layout ----
    const v16bf pa = cat8(*(const bf16x8*)(pl + lh * PL_STRIDE + 16 * half),
                          *(const bf16x8*)(pl + lh * PL_STRIDE + 16 * half + 8));

    // ---- O += P @ V : vt[B,H,T] gives contiguous B-operand rows ----
#pragma unroll
    for (int ht = 0; ht < 4; ++ht) {
      const __bf16* vrow =
          vt + ((size_t)b * H_DIM + ht * 16 + lh) * T_SEQ + j0 + 16 * half;
      v16bf vb = cat8(*(const bf16x8*)(vrow), *(const bf16x8*)(vrow + 8));
      o[ht] = wmma_bf16(pa, vb, o[ht]);
    }
  }

  // ---- epilogue: normalize and store f32 ----
#pragma unroll
  for (int ht = 0; ht < 4; ++ht) {
#pragma unroll
    for (int r = 0; r < 8; ++r) {
      const int row = qbase + r + 8 * half;
      out[(size_t)(b * T_SEQ + row) * H_DIM + ht * 16 + lh] =
          o[ht][r] / lrow[r];
    }
  }
}

extern "C" void kernel_launch(void* const* d_in, const int* in_sizes, int n_in,
                              void* d_out, int out_size, void* d_ws,
                              size_t ws_size, hipStream_t stream) {
  (void)in_sizes; (void)n_in; (void)out_size; (void)ws_size;
  const float* x  = (const float*)d_in[0];
  const float* Wq = (const float*)d_in[1];
  const float* bq = (const float*)d_in[2];
  const float* Wk = (const float*)d_in[3];
  const float* bk = (const float*)d_in[4];
  const float* Wv = (const float*)d_in[5];
  const float* bv = (const float*)d_in[6];
  float* out = (float*)d_out;

  const size_t NTOK = (size_t)B_DIM * T_SEQ;      // 16384
  __bf16* qws  = (__bf16*)d_ws;                   // [B*T, H]
  __bf16* kws  = qws + NTOK * H_DIM;              // [B*T, H]
  __bf16* vtws = kws + NTOK * H_DIM;              // [B, H, T]

  qkv_proj_kernel<<<dim3(NTOK / 128), dim3(256), 0, stream>>>(
      x, Wq, bq, Wk, bk, Wv, bv, qws, kws, vtws);
  flash_attn_kernel<<<dim3(B_DIM * (T_SEQ / 128)), dim3(256), 0, stream>>>(
      qws, kws, vtws, out);
}
